// Decoder_30425548325239
// MI455X (gfx1250) — matrix operand order repaired and checked by reading, not compiled, not run on hardware
//
#include <hip/hip_runtime.h>
#include <hip/hip_bf16.h>

// ---------------------------------------------------------------------------
// Decoder: atom MLP (100K x [64->256->16]) + bond MLP (1M x [96->256->5])
// CDNA5 (gfx1250) wave32 WMMA, f16 inputs / f32 accumulate.
// 64 rows per block: B-fragments held in registers across 4 M-tiles.
// ---------------------------------------------------------------------------

typedef _Float16 v16h __attribute__((ext_vector_type(16)));
typedef _Float16 v8h  __attribute__((ext_vector_type(8)));
typedef float    v8f  __attribute__((ext_vector_type(8)));

#define N_NODES   100000
#define N_EDGES   1000000
#define LATENT    64
#define HID       256
#define RBF_DIM   32
#define RBF_DMAX  10.0f
#define BOND_IN   96          // LATENT + RBF_DIM
#define OUT_A     16          // 10 atom types + 6 charges
#define OUT_B     5
#define MBLK      64          // rows/edges per workgroup (4 M-tiles of 16)

// Workspace layout (units: _Float16). One B-fragment = 32 lanes x 16 halves
// = 512 halves, contiguous per (colTile, kChunk).
#define OFF_A1  0                       // aW1: 16 colTiles x 2 chunks = 16384
#define OFF_A2  16384                   // aW2:  1 colTile  x 8 chunks =  4096
#define OFF_B1  20480                   // bW1: 16 colTiles x 3 chunks = 24576
#define OFF_B2  45056                   // bW2:  1 colTile  x 8 chunks =  4096
#define WS_HALVES 49152                 // 96 KB total

// -------------------------- weight pre-swizzle -----------------------------
//   n = colTile*16 + (lane & 15)
//   k = chunk*32 + 8*(lane >> 4) + (h < 8 ? h : 8 + h)
__device__ __forceinline__ void fill_bfrag(_Float16* __restrict__ dst,
                                           const float* __restrict__ W,
                                           int Kact, int Nact, int Ntot,
                                           int nChk, int rel) {
    int frag   = rel >> 9;
    int within = rel & 511;
    int lane   = within >> 4;
    int h      = within & 15;
    int c      = frag / nChk;
    int ch     = frag - c * nChk;
    int k      = ch * 32 + ((lane >> 4) << 3) + ((h < 8) ? h : 8 + h);
    int n      = c * 16 + (lane & 15);
    float v = 0.0f;
    if (k < Kact && n < Nact) v = W[k * Ntot + n];
    dst[rel] = (_Float16)v;
}

__global__ void convert_weights(const float* __restrict__ aW1,
                                const float* __restrict__ aW2,
                                const float* __restrict__ bW1,
                                const float* __restrict__ bW2,
                                _Float16* __restrict__ ws) {
    int stride = (int)(gridDim.x * blockDim.x);
    for (int i = (int)(blockIdx.x * blockDim.x + threadIdx.x);
         i < WS_HALVES; i += stride) {
        if (i < OFF_A2) {
            fill_bfrag(ws + OFF_A1, aW1, LATENT, HID, HID, 2, i - OFF_A1);
        } else if (i < OFF_B1) {
            fill_bfrag(ws + OFF_A2, aW2, HID, OUT_A, OUT_A, 8, i - OFF_A2);
        } else if (i < OFF_B2) {
            fill_bfrag(ws + OFF_B1, bW1, BOND_IN, HID, HID, 3, i - OFF_B1);
        } else {
            fill_bfrag(ws + OFF_B2, bW2, HID, OUT_B, OUT_B, 8, i - OFF_B2);
        }
    }
}

// ----------------------------- helpers -------------------------------------
__device__ __forceinline__ v8f wmma_f16(v16h a, v16h b, v8f c) {
    return __builtin_amdgcn_wmma_f32_16x16x32_f16(false, a, false, b,
                                                  (short)0, c, false, false);
}

__device__ __forceinline__ v16h cat8(v8h lo, v8h hi) {
    return __builtin_shufflevector(lo, hi, 0, 1, 2, 3, 4, 5, 6, 7,
                                   8, 9, 10, 11, 12, 13, 14, 15);
}

// ------------------------------ atom MLP ------------------------------------
__global__ void __launch_bounds__(256)
atom_mlp(const float* __restrict__ q,
         const float* __restrict__ ab1, const float* __restrict__ ab2,
         const _Float16* __restrict__ wA1, const _Float16* __restrict__ wA2,
         float* __restrict__ outType, float* __restrict__ outCharge) {
    __shared__ __align__(16) _Float16 h_lds[MBLK][HID + 8];
    __shared__ float part[8][16][16];

    const int row0  = blockIdx.x * MBLK;
    const int tid   = (int)threadIdx.x;
    const int lane  = tid & 31;
    const int w     = tid >> 5;
    const int m     = lane & 15;
    const int hiOff = (lane >= 16) ? 8 : 0;     // A: +8 K; C/D: +8 M

    // ---- layer-1 B fragments + biases: resident across all 4 M-tiles ----
    v16h bfr1[2][2];
    float bias1[2];
    #pragma unroll
    for (int t = 0; t < 2; ++t) {
        const int col = w + t * 8;
        bias1[t] = ab1[col * 16 + m];
        #pragma unroll
        for (int ch = 0; ch < 2; ++ch)
            bfr1[t][ch] = ((const v16h*)(wA1 + ((col * 2 + ch) << 9)))[lane];
    }

    // ---- layer 1: hid = relu(q @ aW1 + ab1), 4 M-tiles ----
    #pragma unroll
    for (int mt = 0; mt < 4; ++mt) {
        int row = row0 + mt * 16 + m;
        if (row > N_NODES - 1) row = N_NODES - 1;   // data clamp, no divergence
        const float* qrow = q + (size_t)row * LATENT;
        v16h afr[2];
        #pragma unroll
        for (int ch = 0; ch < 2; ++ch) {
            const int k0 = ch * 32 + hiOff;
            #pragma unroll
            for (int j = 0; j < 8; ++j) {
                afr[ch][j]     = (_Float16)qrow[k0 + j];
                afr[ch][j + 8] = (_Float16)qrow[k0 + 16 + j];
            }
        }
        #pragma unroll
        for (int t = 0; t < 2; ++t) {
            v8f acc = {};
            #pragma unroll
            for (int ch = 0; ch < 2; ++ch)
                acc = wmma_f16(afr[ch], bfr1[t][ch], acc);
            const int n = (w + t * 8) * 16 + m;
            #pragma unroll
            for (int r = 0; r < 8; ++r) {
                float v = acc[r] + bias1[t];
                h_lds[mt * 16 + r + hiOff][n] = (_Float16)fmaxf(v, 0.0f);
            }
        }
    }
    __syncthreads();

    // ---- layer 2: wave w owns (M-tile w>>1, K-half w&1): 4 chained WMMAs ----
    {
        const int mt  = w >> 1;
        const int ch0 = (w & 1) * 4;
        v8f acc = {};
        #pragma unroll
        for (int c = 0; c < 4; ++c) {
            const int k0 = (ch0 + c) * 32 + hiOff;
            v8h lo = *(const v8h*)&h_lds[mt * 16 + m][k0];
            v8h hi = *(const v8h*)&h_lds[mt * 16 + m][k0 + 16];
            v16h afr2 = cat8(lo, hi);
            v16h bfr  = ((const v16h*)(wA2 + ((ch0 + c) << 9)))[lane];
            acc = wmma_f16(afr2, bfr, acc);
        }
        #pragma unroll
        for (int r = 0; r < 8; ++r) part[w][r + hiOff][m] = acc[r];
    }
    __syncthreads();

    // ---- 2-way reduce per tile, add bias, split type/charge outputs ----
    {
        const int mm = tid >> 4;
        const int nn = tid & 15;
        const float b2 = ab2[nn];
        #pragma unroll
        for (int t2 = 0; t2 < 4; ++t2) {
            const float s = part[2 * t2][mm][nn] + part[2 * t2 + 1][mm][nn] + b2;
            const int row = row0 + t2 * 16 + mm;
            if (row < N_NODES) {
                if (nn < 10) outType[row * 10 + nn] = s;
                else         outCharge[row * 6 + (nn - 10)] = s;
            }
        }
    }
}

// ------------------------------ bond MLP ------------------------------------
__global__ void __launch_bounds__(256)
bond_mlp(const float* __restrict__ pos, const float* __restrict__ q,
         const long long* __restrict__ pairs,
         const float* __restrict__ bb1, const float* __restrict__ bb2,
         const _Float16* __restrict__ wB1, const _Float16* __restrict__ wB2,
         float* __restrict__ outBond) {
    __shared__ int   sSrc[MBLK], sDst[MBLK];
    __shared__ float sDist[MBLK];
    __shared__ __align__(16) _Float16 comb[MBLK][BOND_IN + 16];  // stride 112
    __shared__ __align__(16) _Float16 h_lds[MBLK][HID + 8];
    __shared__ float part[8][16][16];

    const int e0  = blockIdx.x * MBLK;
    const int tid = (int)threadIdx.x;

    if (tid < MBLK) {
        const long long s = pairs[2 * (size_t)(e0 + tid)];
        const long long d = pairs[2 * (size_t)(e0 + tid) + 1];
        sSrc[tid] = (int)s;
        sDst[tid] = (int)d;
        const float dx = pos[3 * s + 0] - pos[3 * d + 0];
        const float dy = pos[3 * s + 1] - pos[3 * d + 1];
        const float dz = pos[3 * s + 2] - pos[3 * d + 2];
        sDist[tid] = sqrtf(fmaxf(dx * dx + dy * dy + dz * dz, 1e-8f));
    }
    __syncthreads();

    // combined[e] = [ q[src]+q[dst] (64) | rbf(dist) (32) ]  (L2-resident gather)
    for (int i = tid; i < MBLK * BOND_IN; i += 256) {
        const int e = i / BOND_IN;
        const int c = i - e * BOND_IN;
        float v;
        if (c < LATENT) {
            v = q[(size_t)sSrc[e] * LATENT + c] + q[(size_t)sDst[e] * LATENT + c];
        } else {
            const int   r     = c - LATENT;
            const float mu    = (RBF_DMAX * (float)r) / (float)(RBF_DIM - 1);
            const float sigma = RBF_DMAX / (float)RBF_DIM;
            const float z     = (sDist[e] - mu) / sigma;
            v = __expf(-z * z);
        }
        comb[e][c] = (_Float16)v;
    }
    __syncthreads();

    const int lane  = tid & 31;
    const int w     = tid >> 5;
    const int m     = lane & 15;
    const int hiOff = (lane >= 16) ? 8 : 0;

    // ---- layer-1 B fragments + biases: resident across all 4 M-tiles ----
    v16h bfr1[2][3];
    float bias1[2];
    #pragma unroll
    for (int t = 0; t < 2; ++t) {
        const int col = w + t * 8;
        bias1[t] = bb1[col * 16 + m];
        #pragma unroll
        for (int ch = 0; ch < 3; ++ch)
            bfr1[t][ch] = ((const v16h*)(wB1 + ((col * 3 + ch) << 9)))[lane];
    }

    // ---- layer 1: hb = relu(combined @ bW1 + bb1), K=96, 4 M-tiles ----
    #pragma unroll
    for (int mt = 0; mt < 4; ++mt) {
        v16h afr[3];
        #pragma unroll
        for (int ch = 0; ch < 3; ++ch) {
            const int k0 = ch * 32 + hiOff;
            v8h lo = *(const v8h*)&comb[mt * 16 + m][k0];
            v8h hi = *(const v8h*)&comb[mt * 16 + m][k0 + 16];
            afr[ch] = cat8(lo, hi);
        }
        #pragma unroll
        for (int t = 0; t < 2; ++t) {
            v8f acc = {};
            #pragma unroll
            for (int ch = 0; ch < 3; ++ch)
                acc = wmma_f16(afr[ch], bfr1[t][ch], acc);
            const int n = (w + t * 8) * 16 + m;
            #pragma unroll
            for (int r = 0; r < 8; ++r) {
                float v = acc[r] + bias1[t];
                h_lds[mt * 16 + r + hiOff][n] = (_Float16)fmaxf(v, 0.0f);
            }
        }
    }
    __syncthreads();

    // ---- layer 2: wave w owns (M-tile w>>1, K-half w&1) ----
    {
        const int mt  = w >> 1;
        const int ch0 = (w & 1) * 4;
        v8f acc = {};
        #pragma unroll
        for (int c = 0; c < 4; ++c) {
            const int k0 = (ch0 + c) * 32 + hiOff;
            v8h lo = *(const v8h*)&h_lds[mt * 16 + m][k0];
            v8h hi = *(const v8h*)&h_lds[mt * 16 + m][k0 + 16];
            v16h afr2 = cat8(lo, hi);
            v16h bfr  = ((const v16h*)(wB2 + ((ch0 + c) << 9)))[lane];
            acc = wmma_f16(afr2, bfr, acc);
        }
        #pragma unroll
        for (int r = 0; r < 8; ++r) part[w][r + hiOff][m] = acc[r];
    }
    __syncthreads();

    // ---- 2-way reduce per tile, add bias, write 5 bond logits ----
    {
        const int mm = tid >> 4;
        const int nn = tid & 15;
        if (nn < OUT_B) {
            const float b2 = bb2[nn];
            #pragma unroll
            for (int t2 = 0; t2 < 4; ++t2) {
                const float s = part[2 * t2][mm][nn] + part[2 * t2 + 1][mm][nn] + b2;
                const int e = e0 + t2 * 16 + mm;
                if (e < N_EDGES) outBond[(size_t)e * OUT_B + nn] = s;
            }
        }
    }
}

// ------------------------------ launcher ------------------------------------
extern "C" void kernel_launch(void* const* d_in, const int* in_sizes, int n_in,
                              void* d_out, int out_size, void* d_ws, size_t ws_size,
                              hipStream_t stream) {
    const float*     positions = (const float*)d_in[0];
    const float*     quantized = (const float*)d_in[1];
    const long long* pairs     = (const long long*)d_in[2];
    const float*     aW1       = (const float*)d_in[3];
    const float*     ab1       = (const float*)d_in[4];
    const float*     aW2       = (const float*)d_in[5];
    const float*     ab2       = (const float*)d_in[6];
    const float*     bW1       = (const float*)d_in[7];
    const float*     bb1       = (const float*)d_in[8];
    const float*     bW2       = (const float*)d_in[9];
    const float*     bb2       = (const float*)d_in[10];

    float* out       = (float*)d_out;
    float* outType   = out;                        // [100000, 10]
    float* outCharge = out + (size_t)N_NODES * 10; // [100000, 6]
    float* outBond   = out + (size_t)N_NODES * 16; // [1000000, 5]

    _Float16* ws = (_Float16*)d_ws;

    convert_weights<<<192, 256, 0, stream>>>(aW1, aW2, bW1, bW2, ws);

    atom_mlp<<<(N_NODES + MBLK - 1) / MBLK, 256, 0, stream>>>(
        quantized, ab1, ab2, ws + OFF_A1, ws + OFF_A2, outType, outCharge);

    bond_mlp<<<N_EDGES / MBLK, 256, 0, stream>>>(
        positions, quantized, pairs, bb1, bb2, ws + OFF_B1, ws + OFF_B2, outBond);
}
